// Neural_Network_61375082660002
// MI455X (gfx1250) — matrix operand order repaired and checked by reading, not compiled, not run on hardware
//
#include <hip/hip_runtime.h>
#include <hip/hip_bf16.h>
#include <math.h>

typedef __bf16 bf16;
typedef __attribute__((ext_vector_type(16))) __bf16 v16bf;
typedef __attribute__((ext_vector_type(8)))  float  v8f;

#define LSTRIDE 40         // bf16 elems per LDS row: 32 data + 8 pad = 80B (16B aligned, bank-skewed)
#define TBUF (128 * LSTRIDE)  // one tile buffer, in bf16 elements

union FragAB { v16bf v; uint4 u[2]; };
union Pack16 { bf16 h[16]; uint4 u[2]; };

__device__ __forceinline__ void cvt_store16(bf16* dst, const float4 r[4])
{
    Pack16 t;
#pragma unroll
    for (int q = 0; q < 4; q++) {
        t.h[q * 4 + 0] = (bf16)r[q].x; t.h[q * 4 + 1] = (bf16)r[q].y;
        t.h[q * 4 + 2] = (bf16)r[q].z; t.h[q * 4 + 3] = (bf16)r[q].w;
    }
    ((uint4*)dst)[0] = t.u[0];
    ((uint4*)dst)[1] = t.u[1];
}

// C = relu(A @ W^T + bias), A:[M,K] (f32 or bf16, row-major), W:[Nout,K] f32 row-major,
// out: bf16 [M, Nout]. WG tile 128x128, K-step 32, double-buffered LDS, 8 waves x (4x2) WMMA.
template <bool A_F32>
__global__ __launch_bounds__(256) void gemm_bias_relu(
    const void* __restrict__ Ap, const float* __restrict__ W,
    const float* __restrict__ bias, bf16* __restrict__ out, int K, int Nout)
{
    __shared__ bf16 ldsA[2 * TBUF];
    __shared__ bf16 ldsB[2 * TBUF];

    const int tid   = threadIdx.x;
    const int lane  = tid & 31;
    const int wid   = tid >> 5;
    const int waveM = wid & 1;   // 2 waves along M (64 rows each)
    const int waveN = wid >> 1;  // 4 waves along N (32 cols each)
    const int nBase = blockIdx.x * 128;
    const int mBase = blockIdx.y * 128;

    const int lr = tid >> 1;  // loader row 0..127
    const int lh = tid & 1;   // loader half (16 cols)

    const float* Af = (const float*)Ap;
    const bf16*  Ab = (const bf16*)Ap;
    const int aRowOff = (mBase + lr) * K + lh * 16;  // 32-bit indices (max ~8.4M)
    const int wRowOff = (nBase + lr) * K + lh * 16;
    const int stOff   = lr * LSTRIDE + lh * 16;      // LDS staging offset (element index)

    v8f acc[4][2];
    v8f zero = {};
#pragma unroll
    for (int i = 0; i < 4; i++)
#pragma unroll
        for (int j = 0; j < 2; j++) acc[i][j] = zero;

    // Fragment offsets (per ISA 16-bit A(16x32) / B(32x16) VGPR layouts)
    const int rsel  = lane & 15;
    const int aKoff = (lane < 16) ? 0 : 8;   // A: lanes 16-31 hold K+8 / K+24 halves
    const int bKoff = (lane < 16) ? 0 : 8;   // B: lanes 16-31 hold K+8 / K+24 halves
    const int aFragOff = (waveM * 64 + rsel) * LSTRIDE + aKoff;
    const int bFragOff = (waveN * 32 + rsel) * LSTRIDE + bKoff;

    // ---------------- prologue: stage tile 0 into buffer 0 ----------------
    {
        float4 wr[4];
        const float4* ws4 = (const float4*)(W + wRowOff);
#pragma unroll
        for (int q = 0; q < 4; q++) wr[q] = ws4[q];
        if (A_F32) {
            float4 ar[4];
            const float4* as4 = (const float4*)(Af + aRowOff);
#pragma unroll
            for (int q = 0; q < 4; q++) ar[q] = as4[q];
            cvt_store16(&ldsA[stOff], ar);
        } else {
            const uint4* as4 = (const uint4*)(Ab + aRowOff);
            uint4 a0 = as4[0], a1 = as4[1];
            ((uint4*)&ldsA[stOff])[0] = a0;
            ((uint4*)&ldsA[stOff])[1] = a1;
        }
        cvt_store16(&ldsB[stOff], wr);
    }
    __syncthreads();

    const int nT = K >> 5;
    for (int t = 0; t < nT; ++t) {
        const int curOff = (t & 1) * TBUF;
        const int nxtOff = curOff ^ TBUF;

        // ---- issue next tile's global loads first (latency hidden by WMMA) ----
        float4 ar[4], wr[4];
        uint4  arb[2];
        const bool havePf = (t + 1 < nT);
        if (havePf) {
            const int kn = (t + 1) << 5;
            const float4* ws4 = (const float4*)(W + wRowOff + kn);
#pragma unroll
            for (int q = 0; q < 4; q++) wr[q] = ws4[q];
            if (A_F32) {
                const float4* as4 = (const float4*)(Af + aRowOff + kn);
#pragma unroll
                for (int q = 0; q < 4; q++) ar[q] = as4[q];
            } else {
                const uint4* as4 = (const uint4*)(Ab + aRowOff + kn);
                arb[0] = as4[0];
                arb[1] = as4[1];
            }
        }

        // ---- fragments + 8 WMMAs on current buffer ----
        FragAB aF[4], bF[2];
#pragma unroll
        for (int mi = 0; mi < 4; mi++) {
            const bf16* p = &ldsA[curOff + aFragOff + mi * (16 * LSTRIDE)];
            aF[mi].u[0] = *(const uint4*)(p);       // K = aKoff .. aKoff+7
            aF[mi].u[1] = *(const uint4*)(p + 16);  // K = aKoff+16 .. aKoff+23
        }
#pragma unroll
        for (int ni = 0; ni < 2; ni++) {
            const bf16* p = &ldsB[curOff + bFragOff + ni * (16 * LSTRIDE)];
            bF[ni].u[0] = *(const uint4*)(p);       // K = bKoff .. +7
            bF[ni].u[1] = *(const uint4*)(p + 16);  // K = bKoff+16 .. +23
        }
#pragma unroll
        for (int mi = 0; mi < 4; mi++)
#pragma unroll
            for (int ni = 0; ni < 2; ni++)
                acc[mi][ni] = __builtin_amdgcn_wmma_f32_16x16x32_bf16(
                    false, aF[mi].v, false, bF[ni].v, (short)0, acc[mi][ni], false, false);

        // ---- drain prefetch into the other buffer (ds_store path) ----
        if (havePf) {
            if (A_F32) {
                cvt_store16(&ldsA[nxtOff + stOff], ar);
            } else {
                ((uint4*)&ldsA[nxtOff + stOff])[0] = arb[0];
                ((uint4*)&ldsA[nxtOff + stOff])[1] = arb[1];
            }
            cvt_store16(&ldsB[nxtOff + stOff], wr);
        }
        __syncthreads();
    }

    // ---- epilogue: bias + relu + bf16 store (C layout: VGPR r -> M=r / r+8) ----
    const int mAdd = (lane < 16) ? 0 : 8;
#pragma unroll
    for (int mi = 0; mi < 4; mi++) {
#pragma unroll
        for (int ni = 0; ni < 2; ni++) {
            const int n = nBase + waveN * 32 + ni * 16 + rsel;
            const float bv = bias[n];
#pragma unroll
            for (int r = 0; r < 8; r++) {
                const int m = mBase + waveM * 64 + mi * 16 + r + mAdd;
                float v = acc[mi][ni][r] + bv;
                v = v > 0.0f ? v : 0.0f;
                out[m * Nout + n] = (bf16)v;
            }
        }
    }
}

// logits = h2 @ W3^T + b3 ; sum log_softmax(logits)[Y]  (one wave per row)
__global__ __launch_bounds__(256) void head_logsoftmax(
    const bf16* __restrict__ h2, const float* __restrict__ W3,
    const float* __restrict__ b3, const int* __restrict__ Y,
    int rowBase, float* __restrict__ ll)
{
    __shared__ float w3s[10 * 2048];
    for (int i = threadIdx.x; i < 10 * 2048; i += 256) w3s[i] = W3[i];
    __syncthreads();

    const int wid  = threadIdx.x >> 5;
    const int lane = threadIdx.x & 31;
    const int localRow = blockIdx.x * 8 + wid;
    const bf16* hrow = h2 + localRow * 2048;

    float acc[10];
#pragma unroll
    for (int c = 0; c < 10; c++) acc[c] = 0.0f;

    for (int k = lane; k < 2048; k += 32) {
        const float hv = (float)hrow[k];
#pragma unroll
        for (int c = 0; c < 10; c++) acc[c] += hv * w3s[c * 2048 + k];
    }
#pragma unroll
    for (int c = 0; c < 10; c++) {
#pragma unroll
        for (int off = 16; off >= 1; off >>= 1) acc[c] += __shfl_down(acc[c], off, 32);
    }
    if (lane == 0) {
        float logits[10];
        float mx = -1e30f;
#pragma unroll
        for (int c = 0; c < 10; c++) { logits[c] = acc[c] + b3[c]; mx = fmaxf(mx, logits[c]); }
        float s = 0.0f;
#pragma unroll
        for (int c = 0; c < 10; c++) s += expf(logits[c] - mx);
        const int y = Y[rowBase + localRow];
        atomicAdd(ll, logits[y] - (mx + logf(s)));
    }
}

__global__ void sqsum_kernel(const float* __restrict__ p, int n, float* __restrict__ accum)
{
    float s = 0.0f;
    for (int i = blockIdx.x * blockDim.x + threadIdx.x; i < n; i += gridDim.x * blockDim.x) {
        const float v = p[i];
        s += v * v;
    }
#pragma unroll
    for (int off = 16; off >= 1; off >>= 1) s += __shfl_down(s, off, 32);
    if ((threadIdx.x & 31) == 0) atomicAdd(accum, s);
}

__global__ void init_accum(float* a) { a[0] = 0.0f; a[1] = 0.0f; }

__global__ void finalize_kernel(const float* a, float* out)
{
    const double ll = (double)a[0];
    const double sq = (double)a[1];
    // d = 6,316,042 params ; log(2*pi*100) = 6.443047252397436
    const double lp = -0.5 * (sq / 100.0 + 6316042.0 * 6.4430472523974365);
    out[0] = (float)(ll + lp);
}

extern "C" void kernel_launch(void* const* d_in, const int* in_sizes, int n_in,
                              void* d_out, int out_size, void* d_ws, size_t ws_size,
                              hipStream_t stream)
{
    (void)in_sizes; (void)n_in; (void)out_size; (void)ws_size;
    const float* X  = (const float*)d_in[0];
    const int*   Y  = (const int*)d_in[1];
    const float* W1 = (const float*)d_in[2];
    const float* b1 = (const float*)d_in[3];
    const float* W2 = (const float*)d_in[4];
    const float* b2 = (const float*)d_in[5];
    const float* W3 = (const float*)d_in[6];
    const float* b3 = (const float*)d_in[7];
    float* out = (float*)d_out;

    const int N = 16384, IN = 1024, H = 2048, CHUNK = 4096;

    bf16* h1 = (bf16*)d_ws;                       // [CHUNK, H] bf16 : 16 MB
    bf16* h2 = h1 + (size_t)CHUNK * H;            // [CHUNK, H] bf16 : 16 MB
    float* accum = (float*)(h2 + (size_t)CHUNK * H);  // [0]=log_lik, [1]=sq

    init_accum<<<1, 1, 0, stream>>>(accum);
    sqsum_kernel<<<256, 256, 0, stream>>>(W1, H * IN, accum + 1);
    sqsum_kernel<<<4,   256, 0, stream>>>(b1, H,      accum + 1);
    sqsum_kernel<<<256, 256, 0, stream>>>(W2, H * H,  accum + 1);
    sqsum_kernel<<<4,   256, 0, stream>>>(b2, H,      accum + 1);
    sqsum_kernel<<<16,  256, 0, stream>>>(W3, 10 * H, accum + 1);
    sqsum_kernel<<<1,   32,  0, stream>>>(b3, 10,     accum + 1);

    for (int c0 = 0; c0 < N; c0 += CHUNK) {
        gemm_bias_relu<true><<<dim3(H / 128, CHUNK / 128), 256, 0, stream>>>(
            X + (size_t)c0 * IN, W1, b1, h1, IN, H);
        gemm_bias_relu<false><<<dim3(H / 128, CHUNK / 128), 256, 0, stream>>>(
            h1, W2, b2, h2, H, H);
        head_logsoftmax<<<CHUNK / 8, 256, 0, stream>>>(h2, W3, b3, Y, c0, accum);
    }
    finalize_kernel<<<1, 1, 0, stream>>>(accum, out);
}
